// HeteroGraphSAGE_84920093377265
// MI455X (gfx1250) — compile-verified, matrix operand order checked by reading, not run on hardware
//
#include <hip/hip_runtime.h>

typedef __attribute__((ext_vector_type(16))) __bf16 v16bf;
typedef __attribute__((ext_vector_type(16))) unsigned short v16u;
typedef __attribute__((ext_vector_type(8)))  float v8f;
typedef unsigned short ushort_t;

static __device__ __forceinline__ ushort_t f2bf(float f) {
    union { float f; unsigned u; } x; x.f = f;
    unsigned u = x.u;
    unsigned r = u + 0x7FFFu + ((u >> 16) & 1u);   // round-to-nearest-even
    return (ushort_t)(r >> 16);
}

// ---------------------------------------------------------------------------
// zero a float4 region
__global__ void zero_f4(float4* __restrict__ p, int n4) {
    int i = blockIdx.x * blockDim.x + threadIdx.x;
    if (i < n4) p[i] = float4{0.f, 0.f, 0.f, 0.f};
}

// ---------------------------------------------------------------------------
// degree: one thread per edge
__global__ void degree_kernel(const int* __restrict__ dst, float* __restrict__ deg, int E) {
    int e = blockIdx.x * blockDim.x + threadIdx.x;
    if (e < E)
        __hip_atomic_fetch_add(&deg[dst[e]], 1.0f, __ATOMIC_RELAXED, __HIP_MEMORY_SCOPE_AGENT);
}

// ---------------------------------------------------------------------------
// weighted scatter: one wave32 per edge, lane handles float4 (128 floats/row)
__global__ __launch_bounds__(256)
void scatter_edges(const float* __restrict__ hsrc, const int* __restrict__ src,
                   const int* __restrict__ dst, const float* __restrict__ ew,
                   float* __restrict__ s, int E) {
    int gt  = blockIdx.x * blockDim.x + threadIdx.x;
    int wid = gt >> 5;
    int lane = gt & 31;
    if (wid >= E) return;
    int se = src[wid];
    int de = dst[wid];
    float w = ew[wid];
    float4 v = ((const float4*)(hsrc + (size_t)se * 128))[lane];
    float* dp = s + (size_t)de * 128 + lane * 4;
    __hip_atomic_fetch_add(dp + 0, v.x * w, __ATOMIC_RELAXED, __HIP_MEMORY_SCOPE_AGENT);
    __hip_atomic_fetch_add(dp + 1, v.y * w, __ATOMIC_RELAXED, __HIP_MEMORY_SCOPE_AGENT);
    __hip_atomic_fetch_add(dp + 2, v.z * w, __ATOMIC_RELAXED, __HIP_MEMORY_SCOPE_AGENT);
    __hip_atomic_fetch_add(dp + 3, v.w * w, __ATOMIC_RELAXED, __HIP_MEMORY_SCOPE_AGENT);
}

// ---------------------------------------------------------------------------
// pre-swizzle combined weight [Wself;Wneigh] (K=256, N=128) f32 -> bf16 WMMA
// B-fragment layout: Wsw[(ntile*8 + kstep)*32 + lane][16] (32B contiguous/lane)
// 16-bit operand layout: lane = (n&15) + 16*khalf; elem 2r+j -> K = kb(r,khalf)+j
__global__ __launch_bounds__(256)
void pack_weights(const float* __restrict__ Wself, const float* __restrict__ Wneigh,
                  ushort_t* __restrict__ Wsw) {
    int tid = blockIdx.x * blockDim.x + threadIdx.x;   // 2048 fragments
    if (tid >= 2048) return;
    int lane  = tid & 31;
    int kstep = (tid >> 5) & 7;
    int ntile = tid >> 8;                              // 0..7
    int col   = ntile * 16 + (lane & 15);
    int khalf = lane >> 4;
    ushort_t frag[16];
#pragma unroll
    for (int r = 0; r < 8; ++r) {
        int kb = ((r < 4) ? (2 * r + 8 * khalf) : (16 + 2 * (r - 4) + 8 * khalf)) + kstep * 32;
#pragma unroll
        for (int j = 0; j < 2; ++j) {
            int K = kb + j;
            float v = (K < 128) ? Wself[(size_t)K * 128 + col]
                                : Wneigh[(size_t)(K - 128) * 128 + col];
            frag[2 * r + j] = f2bf(v);
        }
    }
    v16u out;
#pragma unroll
    for (int i = 0; i < 16; ++i) out[i] = frag[i];
    *(v16u*)(Wsw + ((size_t)(ntile * 8 + kstep) * 32 + lane) * 16) = out;
}

// ---------------------------------------------------------------------------
// fused SAGE dense part:  out = [h_dst | s/deg] @ [Wself;Wneigh]_bf16 + b
// block = 256 (8 waves) handles a 16-node M-tile; wave w computes N-tile w.
// A fragments (shared by all 8 waves) staged in LDS in WMMA layout.
__global__ __launch_bounds__(256)
void combine_gemm(const float* __restrict__ xdst, const float* __restrict__ s,
                  const float* __restrict__ deg, const ushort_t* __restrict__ Wsw,
                  const float* __restrict__ bias, float* __restrict__ out,
                  int n, int accumulate, int relu) {
    __shared__ __attribute__((aligned(32))) ushort_t Afrag[8][32][16];   // 8 KB

    const int tid   = threadIdx.x;
    const int lane  = tid & 31;
    const int wave  = tid >> 5;
    const int row   = lane & 15;
    const int khalf = lane >> 4;

    int node = blockIdx.x * 16 + row;
    if (node >= n) node = n - 1;
    const float* xr = xdst + (size_t)node * 128;
    const float* sr = s    + (size_t)node * 128;
    const float inv = 1.0f / fmaxf(deg[node], 1.0f);

    // stage: thread (kstep=wave, lane) builds its 16-element bf16 A fragment
    {
        const int kstep = wave;
        ushort_t* dp = &Afrag[kstep][lane][0];
#pragma unroll
        for (int r = 0; r < 8; ++r) {
            int kb = ((r < 4) ? (2 * r + 8 * khalf) : (16 + 2 * (r - 4) + 8 * khalf)) + kstep * 32;
#pragma unroll
            for (int j = 0; j < 2; ++j) {
                int K = kb + j;
                float v = (K < 128) ? xr[K] : sr[K - 128] * inv;
                dp[2 * r + j] = f2bf(v);
            }
        }
    }
    __syncthreads();

    v8f acc = {};
    const ushort_t* wp = Wsw + (size_t)wave * 8 * 32 * 16;   // this wave's N-tile
#pragma unroll
    for (int k = 0; k < 8; ++k) {                            // K = 256 in 8 steps of 32
        v16u au = *(const v16u*)&Afrag[k][lane][0];
        v16u bu = *(const v16u*)(wp + (size_t)(k * 32 + lane) * 16);
        v16bf a = __builtin_bit_cast(v16bf, au);
        v16bf b = __builtin_bit_cast(v16bf, bu);
        acc = __builtin_amdgcn_wmma_f32_16x16x32_bf16(false, a, false, b,
                                                      (short)0, acc, false, false);
    }

    // epilogue: C/D layout -> lane holds col (lane&15)+16*wave, rows r+8*(lane>>4)
    const int col = wave * 16 + (lane & 15);
    const float bv = bias[col];
#pragma unroll
    for (int r = 0; r < 8; ++r) {
        int m  = r + 8 * (lane >> 4);
        int n2 = blockIdx.x * 16 + m;
        if (n2 < n) {
            size_t idx = (size_t)n2 * 128 + col;
            float v = acc[r] + bv;
            if (accumulate) v += out[idx];
            if (relu) v = fmaxf(v, 0.0f);
            out[idx] = v;
        }
    }
}

// ---------------------------------------------------------------------------
// classifier: wave32 per node, out[n,2] = h[n,:]@Wc + bc
__global__ __launch_bounds__(256)
void classify_kernel(const float* __restrict__ h, const float* __restrict__ Wc,
                     const float* __restrict__ bc, float* __restrict__ out, int n) {
    int gt  = blockIdx.x * blockDim.x + threadIdx.x;
    int wid = gt >> 5;
    int lane = gt & 31;
    if (wid >= n) return;
    float4 hv = ((const float4*)(h + (size_t)wid * 128))[lane];
    float e[4] = {hv.x, hv.y, hv.z, hv.w};
    float a0 = 0.f, a1 = 0.f;
#pragma unroll
    for (int j = 0; j < 4; ++j) {
        int k = lane * 4 + j;
        a0 += e[j] * Wc[k * 2 + 0];
        a1 += e[j] * Wc[k * 2 + 1];
    }
#pragma unroll
    for (int off = 16; off > 0; off >>= 1) {
        a0 += __shfl_down(a0, off, 32);
        a1 += __shfl_down(a1, off, 32);
    }
    if (lane == 0) {
        out[(size_t)wid * 2 + 0] = a0 + bc[0];
        out[(size_t)wid * 2 + 1] = a1 + bc[1];
    }
}

// ---------------------------------------------------------------------------
static inline int cdiv(long long a, long long b) { return (int)((a + b - 1) / b); }

extern "C" void kernel_launch(void* const* d_in, const int* in_sizes, int n_in,
                              void* d_out, int out_size, void* d_ws, size_t ws_size,
                              hipStream_t stream) {
    const float* x_app   = (const float*)d_in[0];
    const float* x_attr  = (const float*)d_in[1];
    const float* ew0     = (const float*)d_in[2];
    const float* ew1     = (const float*)d_in[3];
    const float* ew2     = (const float*)d_in[4];
    const float* Wself1  = (const float*)d_in[5];   // [3,128,128]
    const float* Wneigh1 = (const float*)d_in[6];
    const float* b1      = (const float*)d_in[7];   // [3,128]
    const float* Wself2  = (const float*)d_in[8];
    const float* Wneigh2 = (const float*)d_in[9];
    const float* b2      = (const float*)d_in[10];
    const float* Wc      = (const float*)d_in[11];  // [128,2]
    const float* bc      = (const float*)d_in[12];  // [2]
    const int* src0 = (const int*)d_in[13];
    const int* dst0 = (const int*)d_in[14];
    const int* src1 = (const int*)d_in[15];
    const int* dst1 = (const int*)d_in[16];
    const int* src2 = (const int*)d_in[17];
    const int* dst2 = (const int*)d_in[18];

    const int NAPP  = in_sizes[0] / 128;
    const int NATTR = in_sizes[1] / 128;
    const int E0 = in_sizes[2], E1 = in_sizes[3], E2 = in_sizes[4];
    const int NMAX = (NAPP > NATTR) ? NAPP : NATTR;

    // ---- workspace layout (floats) ----
    auto pad64 = [](size_t x) { return (x + 63) & ~size_t(63); };
    float* ws = (float*)d_ws;
    size_t o = 0;
    float* deg0   = ws + o; o += pad64(NATTR);
    float* deg1   = ws + o; o += pad64(NAPP);
    float* deg2   = ws + o; o += pad64(NAPP);
    size_t degTot = o;                       // contiguous deg region (mult of 64)
    float* sbuf   = ws + o; o += (size_t)NMAX * 128;
    float* h1attr = ws + o; o += (size_t)NATTR * 128;
    float* h1app  = ws + o; o += (size_t)NAPP * 128;
    float* h2app  = ws + o; o += (size_t)NAPP * 128;
    ushort_t* wsw = (ushort_t*)(ws + o);     // 5 * 2048 frags * 16 ushorts
    const size_t WCOMBO = 2048 * 16;         // ushorts per combo

    const int B = 256;

    // ---- degrees (shared by both layers) ----
    zero_f4<<<cdiv(degTot / 4, B), B, 0, stream>>>((float4*)deg0, (int)(degTot / 4));
    degree_kernel<<<cdiv(E0, B), B, 0, stream>>>(dst0, deg0, E0);
    degree_kernel<<<cdiv(E1, B), B, 0, stream>>>(dst1, deg1, E1);
    degree_kernel<<<cdiv(E2, B), B, 0, stream>>>(dst2, deg2, E2);

    // ---- pack weights to bf16 WMMA fragment layout (5 used combos) ----
    pack_weights<<<8, B, 0, stream>>>(Wself1 + 0 * 16384, Wneigh1 + 0 * 16384, wsw + 0 * WCOMBO);
    pack_weights<<<8, B, 0, stream>>>(Wself1 + 1 * 16384, Wneigh1 + 1 * 16384, wsw + 1 * WCOMBO);
    pack_weights<<<8, B, 0, stream>>>(Wself1 + 2 * 16384, Wneigh1 + 2 * 16384, wsw + 2 * WCOMBO);
    pack_weights<<<8, B, 0, stream>>>(Wself2 + 1 * 16384, Wneigh2 + 1 * 16384, wsw + 3 * WCOMBO);
    pack_weights<<<8, B, 0, stream>>>(Wself2 + 2 * 16384, Wneigh2 + 2 * 16384, wsw + 4 * WCOMBO);

    // ---- one relation pass: zero-s, scatter, fused GEMM ----
    auto run_sage = [&](const float* hsrc, const float* hdst, const int* srcI,
                        const int* dstI, const float* ewI, int Ep, const float* degI,
                        int ndst, const ushort_t* WswI, const float* biasI,
                        float* outI, int accum, int relu) {
        int n4 = (ndst * 128) / 4;
        zero_f4<<<cdiv(n4, B), B, 0, stream>>>((float4*)sbuf, n4);
        scatter_edges<<<cdiv((long long)Ep * 32, B), B, 0, stream>>>(hsrc, srcI, dstI, ewI, sbuf, Ep);
        combine_gemm<<<cdiv(ndst, 16), B, 0, stream>>>(hdst, sbuf, degI, WswI, biasI,
                                                       outI, ndst, accum, relu);
    };

    // ---- layer 1 ----
    run_sage(x_app,  x_attr, src0, dst0, ew0, E0, deg0, NATTR, wsw + 0 * WCOMBO, b1 + 0,   h1attr, 0, 1);
    run_sage(x_attr, x_app,  src1, dst1, ew1, E1, deg1, NAPP,  wsw + 1 * WCOMBO, b1 + 128, h1app,  0, 0);
    run_sage(x_app,  x_app,  src2, dst2, ew2, E2, deg2, NAPP,  wsw + 2 * WCOMBO, b1 + 256, h1app,  1, 1);

    // ---- layer 2 (attr output unused by classifier -> skip relation 0) ----
    run_sage(h1attr, h1app, src1, dst1, ew1, E1, deg1, NAPP, wsw + 3 * WCOMBO, b2 + 128, h2app, 0, 0);
    run_sage(h1app,  h1app, src2, dst2, ew2, E2, deg2, NAPP, wsw + 4 * WCOMBO, b2 + 256, h2app, 1, 0);

    // ---- classifier ----
    classify_kernel<<<cdiv((long long)NAPP * 32, B), B, 0, stream>>>(h2app, Wc, bc, (float*)d_out, NAPP);

    (void)n_in; (void)out_size; (void)ws_size;
}
